// EntropyGuidedAttention_74577812127894
// MI455X (gfx1250) — matrix operand order repaired and checked by reading, not compiled, not run on hardware
//
#include <hip/hip_runtime.h>
#include <hip/hip_bf16.h>

typedef _Float16 half_t;
typedef __attribute__((ext_vector_type(16))) _Float16 v16h;
typedef __attribute__((ext_vector_type(8)))  _Float16 v8h;
typedef __attribute__((ext_vector_type(4)))  _Float16 v4h;
typedef __attribute__((ext_vector_type(8)))  float    v8f;
typedef __attribute__((ext_vector_type(4)))  float    v4f;
typedef __attribute__((ext_vector_type(4)))  unsigned int u32x4;
typedef __attribute__((ext_vector_type(8)))  int      i32x8;
typedef __attribute__((ext_vector_type(4)))  int      i32x4;

// ---------------------------------------------------------------------------
// CDNA5 WMMA helpers (wave32, 16x16x32 f16 -> f32 accum)
// ---------------------------------------------------------------------------
static __device__ __forceinline__ v8f wmma16x16x32(v16h a, v16h b, v8f c) {
  return __builtin_amdgcn_wmma_f32_16x16x32_f16(false, a, false, b, (short)0, c,
                                                false, false);
}

// A fragment (16x32 f16, row-major, leading dim `ld` halves, ld % 8 == 0,
// base 16B-aligned). Per lane: two contiguous 8-half (16B) runs.
static __device__ __forceinline__ v16h load_a_h(const half_t* p, int ld, int lane) {
  const int m = lane & 15;
  const int g = (lane >> 4) * 8;
  const v8h lo = *(const v8h*)(p + m * ld + g);
  const v8h hi = *(const v8h*)(p + m * ld + g + 16);
  v16h f;
#pragma unroll
  for (int i = 0; i < 8; ++i) { f[i] = lo[i]; f[8 + i] = hi[i]; }
  return f;
}

// A fragment from f32 source (masked probs in LDS), converted on the fly.
static __device__ __forceinline__ v16h load_a_f32(const float* p, int ld, int lane) {
  const int m = lane & 15;
  const int g = (lane >> 4) * 8;
  const float* r0 = p + m * ld + g;
  const v4f x0 = *(const v4f*)(r0);
  const v4f x1 = *(const v4f*)(r0 + 4);
  const v4f x2 = *(const v4f*)(r0 + 16);
  const v4f x3 = *(const v4f*)(r0 + 20);
  v16h f;
#pragma unroll
  for (int j = 0; j < 4; ++j) {
    f[j]      = (half_t)x0[j];
    f[4 + j]  = (half_t)x1[j];
    f[8 + j]  = (half_t)x2[j];
    f[12 + j] = (half_t)x3[j];
  }
  return f;
}

// B fragment (32x16) from row-major [k][n] memory; lanes hold N columns,
// halves hold K in table order (k = 8h+i, 16+8h+i).
static __device__ __forceinline__ v16h load_b_kn(const half_t* p, int ld, int lane) {
  const int n = lane & 15;
  const int g = (lane >> 4) * 8;
  v16h f;
#pragma unroll
  for (int i = 0; i < 8; ++i) {
    f[i]     = p[(size_t)(g + i) * ld + n];
    f[8 + i] = p[(size_t)(g + 16 + i) * ld + n];
  }
  return f;
}

// ---------------------------------------------------------------------------
// wave32 reductions / sortable float bits
// ---------------------------------------------------------------------------
static __device__ __forceinline__ float wave_rmax(float v) {
#pragma unroll
  for (int o = 16; o > 0; o >>= 1) v = fmaxf(v, __shfl_xor(v, o, 32));
  return v;
}
static __device__ __forceinline__ float wave_rsum(float v) {
#pragma unroll
  for (int o = 16; o > 0; o >>= 1) v += __shfl_xor(v, o, 32);
  return v;
}
static __device__ __forceinline__ int wave_isum(int v) {
#pragma unroll
  for (int o = 16; o > 0; o >>= 1) v += __shfl_xor(v, o, 32);
  return v;
}
static __device__ __forceinline__ unsigned f2sort(float f) {
  unsigned u = __float_as_uint(f);
  return u ^ ((u >> 31) ? 0xFFFFFFFFu : 0x80000000u);
}

// ---------------------------------------------------------------------------
// Tiled WMMA GEMM: C = X[M][K] @ W^T + bias.  W is [N][K] row-major.
// TRANS=false: Out[m][n] (pitch N).  TRANS=true: Out[n][m] (pitch M) --
// used to produce K^T directly so the attention B fragments vectorize.
// 64x64 block tile, 8 wave32 each doing a 16x32 sub-tile.
// ---------------------------------------------------------------------------
template <typename TO, bool TRANS>
__global__ __launch_bounds__(256)
void gemm_bias_wmma(const float* __restrict__ X, const float* __restrict__ W,
                    const float* __restrict__ bias, TO* __restrict__ Out,
                    int M, int N, int K) {
  __shared__ alignas(16) half_t a_lds[64][40];   // 64 m x 32 k (+pad, 80B pitch)
  __shared__ alignas(16) half_t b_lds[32][72];   // 32 k x 64 n (+pad, 144B pitch)
  const int tid = threadIdx.x, lane = tid & 31, wave = tid >> 5;
  const int bm = blockIdx.y * 64, bn = blockIdx.x * 64;
  const int wm = (wave & 3) * 16;
  const int wn = (wave >> 2) * 32;
  v8f c0 = {}, c1 = {};
  for (int k0 = 0; k0 < K; k0 += 32) {
#pragma unroll
    for (int i = 0; i < 2; ++i) {
      const int idx = tid + i * 256;        // 0..511 float4 chunks
      const int r = idx >> 3;               // 0..63
      const int c4 = (idx & 7) * 4;         // 0..28
      const v4f av = *(const v4f*)&X[(size_t)(bm + r) * K + k0 + c4];
      v4h ah;
#pragma unroll
      for (int j = 0; j < 4; ++j) ah[j] = (half_t)av[j];
      *(v4h*)&a_lds[r][c4] = ah;            // ds_store_b64
      const v4f bv = *(const v4f*)&W[(size_t)(bn + r) * K + k0 + c4];
#pragma unroll
      for (int j = 0; j < 4; ++j) b_lds[c4 + j][r] = (half_t)bv[j];  // transpose
    }
    __syncthreads();
    if (k0 + 32 < K) {  // global_prefetch_b8 for the next K slab
      __builtin_prefetch(&X[(size_t)(bm + (tid >> 2)) * K + k0 + 32], 0, 1);
      __builtin_prefetch(&W[(size_t)(bn + (tid >> 2)) * K + k0 + 32], 0, 1);
    }
    const v16h a  = load_a_h(&a_lds[wm][0], 40, lane);
    const v16h b0 = load_b_kn(&b_lds[0][wn], 72, lane);
    const v16h b1 = load_b_kn(&b_lds[0][wn + 16], 72, lane);
    c0 = wmma16x16x32(a, b0, c0);
    c1 = wmma16x16x32(a, b1, c1);
    __syncthreads();
  }
#pragma unroll
  for (int r = 0; r < 8; ++r) {
    const int m  = bm + wm + r + (lane >> 4) * 8;
    const int n0 = bn + wn + (lane & 15);
    const float y0 = c0[r] + bias[n0];
    const float y1 = c1[r] + bias[n0 + 16];
    if (TRANS) {
      Out[(size_t)n0 * M + m]        = (TO)y0;
      Out[(size_t)(n0 + 16) * M + m] = (TO)y1;
    } else {
      Out[(size_t)m * N + n0]      = (TO)y0;
      Out[(size_t)m * N + n0 + 16] = (TO)y1;
    }
  }
}

// ---------------------------------------------------------------------------
// Fused attention with exact per-row top-k (entropy-gated keep count).
// grid = (L/16 row strips, H heads); block = 256 threads (8 wave32).
// Full 16x2048 f32 score strip stays resident in the 320KB LDS.
// Q tile is staged by the Tensor Data Mover when the builtin is available.
// Kt is the transposed key matrix [D][L] so B fragments load row-major [k][n] (lanes read adjacent keys).
// ---------------------------------------------------------------------------
__global__ __launch_bounds__(256)
void attn_topk_kernel(const half_t* __restrict__ Qh, const half_t* __restrict__ Kt,
                      const half_t* __restrict__ Vh, const float* __restrict__ ema,
                      const float* __restrict__ thr, float* __restrict__ Y,
                      int L, int D) {
  extern __shared__ char smem[];
  float*    S        = (float*)smem;                                         // 16 x L
  half_t*   qls      = (half_t*)(smem + (size_t)16 * L * 4);                 // 16 x 64 (dense)
  unsigned* hist     = (unsigned*)(smem + (size_t)16 * L * 4 + 16 * 64 * 2); // 16 x 256
  float*    pvred    = (float*)hist;                                         // 16 x 64 (reuse)
  float*    rowscale = (float*)(smem + (size_t)16 * L * 4 + 16 * 64 * 2 + 16 * 256 * 4);

  const int tid = threadIdx.x, lane = tid & 31, wave = tid >> 5;
  const int h   = blockIdx.y;
  const int qm0 = blockIdx.x * 16;
  const int hd  = h * 64;

  // Per-head keep count from frozen entropy EMA vs threshold.
  const float e  = ema[h] - thr[h];
  const float sr = 0.1f + 0.9f / (1.0f + __expf(-e));
  int kkeep = (int)((float)L * (1.0f - sr));   // trunc, like astype(int32)
  if (kkeep < 1) kkeep = 1;

  // ---- Stage Q tile (16 x 64 f16, row stride D) into LDS ----
#if defined(__gfx1250__) && __has_builtin(__builtin_amdgcn_tensor_load_to_lds)
  if (wave == 0) {
    // TDM descriptor: 2-D tile, 16 rows x 64 halves, global pitch D halves.
    const unsigned long long ga =
        (unsigned long long)(size_t)(Qh + (size_t)qm0 * D + hd);
    const unsigned lds_off = (unsigned)(size_t)qls;  // LDS byte address (low 32b)
    u32x4 g0;
    g0[0] = 1u;                                   // count=1, user descriptor
    g0[1] = lds_off;                              // lds_addr
    g0[2] = (unsigned)(ga & 0xFFFFFFFFu);         // global_addr[31:0]
    g0[3] = (unsigned)((ga >> 32) & 0x01FFFFFFu)  // global_addr[56:32]
          | (2u << 30);                           // type = 2 ("image")
    i32x8 g1;
    g1[0] = 1 << 16;                              // data_size = 1 (2 bytes)
    g1[1] = (int)((64u & 0xFFFFu) << 16);         // tensor_dim0[15:0] = 64
    g1[2] = (int)((16u & 0xFFFFu) << 16);         // dim0 hi=0 | tensor_dim1 = 16
    g1[3] = (int)(64u << 16);                     // dim1 hi=0 | tile_dim0 = 64
    g1[4] = 16;                                   // tile_dim1 = 16, tile_dim2 = 0
    g1[5] = 1024;                                 // tensor_dim0_stride = D
    g1[6] = 0;
    g1[7] = 0;
    i32x4 z4 = {0, 0, 0, 0};
#if __clang_major__ >= 23
    i32x8 z8 = {0, 0, 0, 0, 0, 0, 0, 0};
    __builtin_amdgcn_tensor_load_to_lds(g0, g1, z4, z4, z8, 0);
#else
    __builtin_amdgcn_tensor_load_to_lds(g0, g1, z4, z4, 0);
#endif
#if __has_builtin(__builtin_amdgcn_s_wait_tensorcnt)
    __builtin_amdgcn_s_wait_tensorcnt(0);
#endif
  }
#else
  {
    const int r = tid >> 4, c4 = (tid & 15) * 4;  // 16 rows x 16 chunks
    *(v4h*)(qls + r * 64 + c4) =
        *(const v4h*)(Qh + (size_t)(qm0 + r) * D + hd + c4);
  }
#endif
  __syncthreads();

  // ---- QK^T: each wave covers key tiles kt = wave*16, +128, ... ----
  const v16h a0 = load_a_h(qls, 64, lane);        // dh 0..31
  const v16h a1 = load_a_h(qls + 32, 64, lane);   // dh 32..63
  const half_t* ktb = Kt + (size_t)hd * L;        // head slice of K^T [64][L]
  for (int kt = wave * 16; kt < L; kt += 128) {
    const v16h b0 = load_b_kn(ktb + kt, L, lane);             // dh rows 0..31
    const v16h b1 = load_b_kn(ktb + (size_t)32 * L + kt, L, lane); // dh 32..63
    v8f c = {};
    c = wmma16x16x32(a0, b0, c);
    c = wmma16x16x32(a1, b1, c);
#pragma unroll
    for (int r = 0; r < 8; ++r) {
      const int m = r + (lane >> 4) * 8;
      S[m * L + kt + (lane & 15)] = c[r] * 0.125f;   // 1/sqrt(64)
    }
  }
  __syncthreads();

  // ---- Per-row softmax stats + exact top-k (2 rows per wave) ----
#pragma unroll 1
  for (int rr = 0; rr < 2; ++rr) {
    const int row = wave * 2 + rr;
    float* Srow = S + (size_t)row * L;

    float mx = -3.4e38f;
    for (int j = lane; j < L; j += 32) mx = fmaxf(mx, Srow[j]);
    mx = wave_rmax(mx);
    float z = 0.f;
    for (int j = lane; j < L; j += 32) z += __expf(Srow[j] - mx);
    z = wave_rsum(z);

    // Radix-select the k-th largest score (exact, on sortable bits).
    unsigned* Hh = hist + row * 256;
    unsigned prefix = 0;
    int remaining = kkeep;
#pragma unroll 1
    for (int shift = 24; shift >= 0; shift -= 8) {
      for (int b = lane; b < 256; b += 32) Hh[b] = 0;
      __syncthreads();   // uniform trip counts across all 8 waves
      const unsigned hm = (shift == 24) ? 0u : (0xFFFFFFFFu << (shift + 8));
      for (int j = lane; j < L; j += 32) {
        const unsigned u = f2sort(Srow[j]);
        if ((u & hm) == prefix) atomicAdd(&Hh[(u >> shift) & 255u], 1u);
      }
      __syncthreads();
      int cum = 0;
      unsigned bsel = 0;
      for (int b = 255; b >= 0; --b) {
        const int cb = (int)Hh[b];
        if (cum + cb >= remaining) { bsel = (unsigned)b; break; }
        cum += cb;
      }
      remaining -= cum;
      prefix |= bsel << shift;
    }
    const unsigned ut = prefix;   // bits of the k-th largest score

    int cgt = 0;
    for (int j = lane; j < L; j += 32) cgt += (f2sort(Srow[j]) > ut) ? 1 : 0;
    cgt = wave_isum(cgt);
    const int ties = kkeep - cgt;

    // In-order pass: keep > threshold plus first `ties` equal entries by index
    // (stable argsort semantics). Overwrite S with masked exp(s - mx).
    int eqseen = 0;
    float ksum = 0.f;
    for (int j0 = 0; j0 < L; j0 += 32) {
      const int j = j0 + lane;
      const float s = Srow[j];
      const unsigned u = f2sort(s);
      const bool eq = (u == ut);
      const unsigned bal = (unsigned)__ballot(eq);
      const int pre = __popc(bal & ((1u << lane) - 1u));
      const bool keep = (u > ut) || (eq && (eqseen + pre) < ties);
      const float p = keep ? __expf(s - mx) : 0.f;
      Srow[j] = p;
      ksum += p;
      eqseen += __popc(bal);
    }
    ksum = wave_rsum(ksum);
    if (lane == 0) rowscale[row] = 1.f / (ksum + 1e-10f * z);
  }
  __syncthreads();

  // ---- PV: waves (dh tile, key half) -> 16x16 f32 tiles; pair-reduce ----
  const int nt = (wave & 3) * 16;
  const int kh = wave >> 2;
  v8f acc = {};
  const int kbeg = kh * (L / 2), kend = kbeg + (L / 2);
  for (int kt = kbeg; kt < kend; kt += 32) {
    const v16h a = load_a_f32(S + kt, L, lane);
    const v16h b = load_b_kn(Vh + (size_t)kt * D + hd + nt, D, lane);
    acc = wmma16x16x32(a, b, acc);
  }
  if (kh == 1) {
#pragma unroll
    for (int r = 0; r < 8; ++r) {
      const int m = r + (lane >> 4) * 8;
      pvred[m * 64 + nt + (lane & 15)] = acc[r];
    }
  }
  __syncthreads();
  if (kh == 0) {
#pragma unroll
    for (int r = 0; r < 8; ++r) {
      const int m = r + (lane >> 4) * 8;
      const int n = nt + (lane & 15);
      const float v = (acc[r] + pvred[m * 64 + n]) * rowscale[m];
      Y[(size_t)(qm0 + m) * D + hd + n] = v;
    }
  }
}

// ---------------------------------------------------------------------------
// Host launcher
// ---------------------------------------------------------------------------
extern "C" void kernel_launch(void* const* d_in, const int* in_sizes, int n_in,
                              void* d_out, int out_size, void* d_ws, size_t ws_size,
                              hipStream_t stream) {
  (void)in_sizes; (void)n_in; (void)out_size; (void)ws_size;
  const float* x   = (const float*)d_in[0];
  const float* Wq  = (const float*)d_in[1];
  const float* bq  = (const float*)d_in[2];
  const float* Wk  = (const float*)d_in[3];
  const float* bk  = (const float*)d_in[4];
  const float* Wv  = (const float*)d_in[5];
  const float* bv  = (const float*)d_in[6];
  const float* Wo  = (const float*)d_in[7];
  const float* bo  = (const float*)d_in[8];
  const float* ema = (const float*)d_in[9];
  const float* thr = (const float*)d_in[10];
  const int L = 2048, D = 1024;

  // Workspace: Q [L][D], Kt [D][L], V [L][D] f16 (3 x 4MB) + Y f32 (8MB).
  half_t* Qh = (half_t*)d_ws;
  half_t* Kt = Qh + (size_t)L * D;
  half_t* Vh = Kt + (size_t)L * D;
  float*  Yf = (float*)(Vh + (size_t)L * D);

  dim3 gg(D / 64, L / 64);   // (16, 32)
  gemm_bias_wmma<half_t, false><<<gg, 256, 0, stream>>>(x, Wq, bq, Qh, L, D, D);
  gemm_bias_wmma<half_t, true ><<<gg, 256, 0, stream>>>(x, Wk, bk, Kt, L, D, D);
  gemm_bias_wmma<half_t, false><<<gg, 256, 0, stream>>>(x, Wv, bv, Vh, L, D, D);

  const size_t smem = (size_t)16 * L * 4   // score strip (128 KB)
                    + 16 * 64 * 2          // Q tile (dense, TDM-compatible)
                    + 16 * 256 * 4         // radix histograms / PV reduce
                    + 16 * 4;              // row scales
  hipFuncSetAttribute((const void*)attn_topk_kernel,
                      hipFuncAttributeMaxDynamicSharedMemorySize, (int)smem);
  attn_topk_kernel<<<dim3(L / 16, 16), 256, smem, stream>>>(Qh, Kt, Vh, ema, thr,
                                                            Yf, L, D);

  gemm_bias_wmma<float, false><<<gg, 256, 0, stream>>>(Yf, Wo, bo, (float*)d_out,
                                                       L, D, D);
}
